// GTLayer_1795296329925
// MI455X (gfx1250) — compile-verified, matrix operand order checked by reading, not run on hardware
//
#include <hip/hip_runtime.h>
#include <hip/hip_bf16.h>

// ---------------------------------------------------------------------------
// GT layer for gfx1250: fused f16-WMMA GEMMs (X->h->q,k,v) + sparse attention
// Weights pre-swizzled to WMMA fragment order -> b128 loads in the GEMM loop.
// ---------------------------------------------------------------------------

typedef __attribute__((ext_vector_type(16))) _Float16 v16h;
typedef __attribute__((ext_vector_type(8)))  _Float16 v8h;
typedef __attribute__((ext_vector_type(8)))  float    v8f;

#define HDIM 128   // hidden size
#define NHEAD 8
#define DHEAD 16
#define MROWS 32   // rows per block (2 m-subtiles of 16)

// ---- order-preserving float <-> uint map for atomicMax on floats ----------
__device__ __forceinline__ unsigned fmap(float f) {
    unsigned u = __float_as_uint(f);
    return (u & 0x80000000u) ? ~u : (u | 0x80000000u);
}
__device__ __forceinline__ float funmap(unsigned u) {
    unsigned b = (u & 0x80000000u) ? (u & 0x7FFFFFFFu) : ~u;
    return __uint_as_float(b);
}

// K-index inside a 16-bit WMMA A/B fragment (CDNA5 ISA 7.12.2):
// lane half (0:lanes0-15, 1:lanes16-31) owns K ranges {0..7,16..23}/{8..15,24..31}
__device__ __forceinline__ int kmap(int t, int half_id) {
    return ((t >> 3) << 4) + (t & 7) + (half_id << 3);
}

// A fragment: two contiguous 16-byte runs of the (LDS) row -> 2x ds_load_b128
__device__ __forceinline__ v16h load_afrag(const _Float16* rowp, int kb, int half_id) {
    v8h lo = *(const v8h*)(rowp + kb + (half_id << 3));
    v8h hi = *(const v8h*)(rowp + kb + 16 + (half_id << 3));
    v16h a;
    #pragma unroll
    for (int t = 0; t < 8; ++t) { a[t] = lo[t]; a[t + 8] = hi[t]; }
    return a;
}

// ---------------------------------------------------------------------------
// Kernel 0: zero d_out, denom; zero mapped-max buffer
// ---------------------------------------------------------------------------
__global__ __launch_bounds__(256) void init_kernel(float* __restrict__ out,
                                                   unsigned* __restrict__ mx,
                                                   float* __restrict__ dn, int N) {
    int idx = blockIdx.x * 256 + threadIdx.x;
    if (idx < N * HDIM) out[idx] = 0.0f;
    if (idx < N * NHEAD) { mx[idx] = 0u; dn[idx] = 0.0f; }
}

// ---------------------------------------------------------------------------
// Kernel 1: convert + swizzle the four 128x128 weight matrices into WMMA
// B-fragment order: wsw[(((mat*8+nt)*4+ks)*32+lane)*16 + t]
//                 = W[ks*32 + kmap(t,lane>>4)][nt*16 + (lane&15)]
// ---------------------------------------------------------------------------
__global__ __launch_bounds__(256) void wswz_kernel(const float* __restrict__ w_in,
                                                   const float* __restrict__ wq,
                                                   const float* __restrict__ wk,
                                                   const float* __restrict__ wv,
                                                   _Float16* __restrict__ wsw) {
    int idx = blockIdx.x * 256 + threadIdx.x;   // 4*16384 threads
    if (idx >= 4 * HDIM * HDIM) return;
    const int t    = idx & 15;
    const int lane = (idx >> 4) & 31;
    const int ks   = (idx >> 9) & 3;
    const int nt   = (idx >> 11) & 7;
    const int mat  = idx >> 14;
    const float* src = (mat == 0) ? w_in : (mat == 1) ? wq : (mat == 2) ? wk : wv;
    const int k = ks * 32 + kmap(t, lane >> 4);
    const int n = nt * 16 + (lane & 15);
    wsw[idx] = (_Float16)src[k * HDIM + n];
}

// ---------------------------------------------------------------------------
// Kernel 2: fused GEMMs with v_wmma_f32_16x16x32_f16.
// Block = 128 threads (4 waves), 32 rows per block (2 m-subtiles).
//   X tile (f32, float4 loads) -> LDS f16 -> h = X@W_in + b (LDS f16)
//   q/k/v = h@W* + b*, q scaled 0.25, stored f32 HEAD-MAJOR: [n][head][d]
// Each B fragment is one 32B contiguous load, reused by 2 WMMAs.
// ---------------------------------------------------------------------------
__global__ __launch_bounds__(128) void hqkv_kernel(
    const float* __restrict__ X,
    const float* __restrict__ b_in, const float* __restrict__ bq,
    const float* __restrict__ bk,   const float* __restrict__ bv,
    const _Float16* __restrict__ wsw,
    float* __restrict__ qf, float* __restrict__ kf, float* __restrict__ vf,
    int N) {
    __shared__ _Float16 sX[MROWS * HDIM];
    __shared__ _Float16 sH[MROWS * HDIM];

    const int tid  = threadIdx.x;
    const int lane = tid & 31;
    const int wave = tid >> 5;
    const int half_id = lane >> 4;
    const int mrow = lane & 15;       // A-frag row this lane owns
    const int ncol = lane & 15;       // B/C/D-frag column this lane owns
    const int row0 = blockIdx.x * MROWS;

    // cooperative load of 32x128 X tile via float4, f32 -> f16
    const float4* Xv = (const float4*)X;          // 32 float4 per row
    for (int idx = tid; idx < MROWS * 32; idx += 128) {
        const int r = idx >> 5, c4 = idx & 31;
        const int row = row0 + r;
        float4 v = (row < N) ? Xv[(size_t)row * 32 + c4]
                             : make_float4(0.f, 0.f, 0.f, 0.f);
        _Float16* p = sX + r * HDIM + c4 * 4;
        p[0] = (_Float16)v.x; p[1] = (_Float16)v.y;
        p[2] = (_Float16)v.z; p[3] = (_Float16)v.w;
    }
    __syncthreads();

    const v16h* wfrag = (const v16h*)wsw;   // fragment-granular view (32B units)

    // ---- h = X @ W_in + b_in : each wave owns 2 of the 8 N-tiles ----------
    #pragma unroll
    for (int ni = 0; ni < 2; ++ni) {
        const int nt = wave * 2 + ni;
        const int n  = nt * 16 + ncol;
        const float bias = b_in[n];
        v8f acc0, acc1;
        #pragma unroll
        for (int r = 0; r < 8; ++r) { acc0[r] = bias; acc1[r] = bias; }
        #pragma unroll
        for (int ks = 0; ks < 4; ++ks) {
            v16h b  = wfrag[((0 * 8 + nt) * 4 + ks) * 32 + lane];
            v16h a0 = load_afrag(sX + mrow * HDIM,        ks * 32, half_id);
            v16h a1 = load_afrag(sX + (16 + mrow) * HDIM, ks * 32, half_id);
            acc0 = __builtin_amdgcn_wmma_f32_16x16x32_f16(
                false, a0, false, b, (short)0, acc0, false, false);
            acc1 = __builtin_amdgcn_wmma_f32_16x16x32_f16(
                false, a1, false, b, (short)0, acc1, false, false);
        }
        #pragma unroll
        for (int r = 0; r < 8; ++r) {
            const int rr = r + (half_id << 3);
            sH[rr * HDIM + n]        = (_Float16)acc0[r];
            sH[(16 + rr) * HDIM + n] = (_Float16)acc1[r];
        }
    }
    __syncthreads();

    // ---- q/k/v = h @ W* + b* ; store head-major f32 -----------------------
    for (int mat = 1; mat <= 3; ++mat) {
        const float* biasp = (mat == 1) ? bq : (mat == 2) ? bk : bv;
        float*       dst   = (mat == 1) ? qf : (mat == 2) ? kf : vf;
        const float scale  = (mat == 1) ? 0.25f : 1.0f;   // DH^-0.5 = 1/4
        #pragma unroll
        for (int ni = 0; ni < 2; ++ni) {
            const int nt = wave * 2 + ni;
            const int c  = nt * 16 + ncol;                // hidden column
            const float bias = biasp[c];
            v8f acc0, acc1;
            #pragma unroll
            for (int r = 0; r < 8; ++r) { acc0[r] = bias; acc1[r] = bias; }
            #pragma unroll
            for (int ks = 0; ks < 4; ++ks) {
                v16h b  = wfrag[((mat * 8 + nt) * 4 + ks) * 32 + lane];
                v16h a0 = load_afrag(sH + mrow * HDIM,        ks * 32, half_id);
                v16h a1 = load_afrag(sH + (16 + mrow) * HDIM, ks * 32, half_id);
                acc0 = __builtin_amdgcn_wmma_f32_16x16x32_f16(
                    false, a0, false, b, (short)0, acc0, false, false);
                acc1 = __builtin_amdgcn_wmma_f32_16x16x32_f16(
                    false, a1, false, b, (short)0, acc1, false, false);
            }
            // column c -> (d = c/8, head = c%8); head-major offset head*16+d
            const int coff = (c & 7) * DHEAD + (c >> 3);
            #pragma unroll
            for (int r = 0; r < 8; ++r) {
                const int row = row0 + r + (half_id << 3);
                if (row < N)
                    dst[(size_t)row * HDIM + coff] = acc0[r] * scale;
                if (row + 16 < N)
                    dst[(size_t)(row + 16) * HDIM + coff] = acc1[r] * scale;
            }
        }
    }
}

// ---------------------------------------------------------------------------
// Kernel 3: SDDMM + running row max. One thread per (edge, head).
// q/k are head-major -> 4x float4 contiguous loads per side (L2-resident).
// ---------------------------------------------------------------------------
__global__ __launch_bounds__(256) void sddmm_kernel(
    const int* __restrict__ rows, const int* __restrict__ cols,
    const float* __restrict__ qf, const float* __restrict__ kf,
    float* __restrict__ sc, unsigned* __restrict__ mx, int E) {
    int idx = blockIdx.x * 256 + threadIdx.x;
    if (idx >= E * NHEAD) return;
    const int e = idx >> 3, head = idx & 7;
    const int i = rows[e], j = cols[e];
    const float4* qp = (const float4*)(qf + (size_t)i * HDIM + head * DHEAD);
    const float4* kp = (const float4*)(kf + (size_t)j * HDIM + head * DHEAD);
    float s = 0.0f;
    #pragma unroll
    for (int t = 0; t < 4; ++t) {
        float4 a = qp[t], b = kp[t];
        s += a.x * b.x + a.y * b.y + a.z * b.z + a.w * b.w;
    }
    sc[idx] = s;
    atomicMax(&mx[i * NHEAD + head], fmap(s));
}

// ---------------------------------------------------------------------------
// Kernel 4: p = exp(score - rowmax); accumulate denominator
// ---------------------------------------------------------------------------
__global__ __launch_bounds__(256) void expsum_kernel(
    const int* __restrict__ rows, float* __restrict__ sc,
    const unsigned* __restrict__ mx, float* __restrict__ dn, int E) {
    int idx = blockIdx.x * 256 + threadIdx.x;
    if (idx >= E * NHEAD) return;
    const int e = idx >> 3, head = idx & 7;
    const int i = rows[e];
    const float p = __expf(sc[idx] - funmap(mx[i * NHEAD + head]));
    sc[idx] = p;
    atomicAdd(&dn[i * NHEAD + head], p);
}

// ---------------------------------------------------------------------------
// Kernel 5: SpMM scatter: out[i, d*8+head] += (p/denom) * v[j, head, d]
// ---------------------------------------------------------------------------
__global__ __launch_bounds__(256) void spmm_kernel(
    const int* __restrict__ rows, const int* __restrict__ cols,
    const float* __restrict__ sc, const float* __restrict__ dn,
    const float* __restrict__ vf, float* __restrict__ out, int E) {
    int idx = blockIdx.x * 256 + threadIdx.x;
    if (idx >= E * NHEAD) return;
    const int e = idx >> 3, head = idx & 7;
    const int i = rows[e], j = cols[e];
    const float a = sc[idx] / dn[i * NHEAD + head];
    const float4* vp = (const float4*)(vf + (size_t)j * HDIM + head * DHEAD);
    float* op = out + (size_t)i * HDIM + head;   // reference layout: d*8 + head
    #pragma unroll
    for (int t = 0; t < 4; ++t) {
        float4 v = vp[t];
        atomicAdd(&op[(t * 4 + 0) * NHEAD], a * v.x);
        atomicAdd(&op[(t * 4 + 1) * NHEAD], a * v.y);
        atomicAdd(&op[(t * 4 + 2) * NHEAD], a * v.z);
        atomicAdd(&op[(t * 4 + 3) * NHEAD], a * v.w);
    }
}

// ---------------------------------------------------------------------------
extern "C" void kernel_launch(void* const* d_in, const int* in_sizes, int n_in,
                              void* d_out, int out_size, void* d_ws, size_t ws_size,
                              hipStream_t stream) {
    const float* X    = (const float*)d_in[0];
    const int*   rows = (const int*)d_in[1];
    const int*   cols = (const int*)d_in[2];
    const float* W_in = (const float*)d_in[3];
    const float* b_in = (const float*)d_in[4];
    const float* Wq   = (const float*)d_in[5];
    const float* bq   = (const float*)d_in[6];
    const float* Wk   = (const float*)d_in[7];
    const float* bk   = (const float*)d_in[8];
    const float* Wv   = (const float*)d_in[9];
    const float* bv   = (const float*)d_in[10];
    float* out = (float*)d_out;

    const int N = in_sizes[0] / HDIM;   // 100000
    const int E = in_sizes[1];          // 1600000

    // workspace carve-up (all 32B-aligned)
    float* qf = (float*)d_ws;                       // N*128 f32 (head-major)
    float* kf = qf + (size_t)N * HDIM;              // N*128
    float* vf = kf + (size_t)N * HDIM;              // N*128
    float* sc = vf + (size_t)N * HDIM;              // E*8 scores -> probs
    unsigned* mx = (unsigned*)(sc + (size_t)E * NHEAD);   // N*8 mapped max
    float* dn = (float*)(mx + (size_t)N * NHEAD);         // N*8 denom
    _Float16* wsw = (_Float16*)(dn + (size_t)N * NHEAD);  // 4*128*128 f16, swizzled

    const int initT = N * HDIM;
    init_kernel<<<(initT + 255) / 256, 256, 0, stream>>>(out, mx, dn, N);
    wswz_kernel<<<(4 * HDIM * HDIM + 255) / 256, 256, 0, stream>>>(W_in, Wq, Wk, Wv, wsw);
    hqkv_kernel<<<(N + MROWS - 1) / MROWS, 128, 0, stream>>>(X, b_in, bq, bk, bv, wsw,
                                                             qf, kf, vf, N);
    const int eT = E * NHEAD;
    sddmm_kernel<<<(eT + 255) / 256, 256, 0, stream>>>(rows, cols, qf, kf, sc, mx, E);
    expsum_kernel<<<(eT + 255) / 256, 256, 0, stream>>>(rows, sc, mx, dn, E);
    spmm_kernel<<<(eT + 255) / 256, 256, 0, stream>>>(rows, cols, sc, dn, vf, out, E);
}
